// Model_TN_81784767250671
// MI455X (gfx1250) — matrix operand order repaired and checked by reading, not compiled, not run on hardware
//
#include <hip/hip_runtime.h>

typedef __attribute__((ext_vector_type(16))) _Float16 v16h;
typedef __attribute__((ext_vector_type(4)))  _Float16 v4h;
typedef __attribute__((ext_vector_type(8)))  float    v8f;
typedef __attribute__((ext_vector_type(2)))  float    f32x2;

static inline long cdivl(long a, long b) { return (a + b - 1) / b; }

// ---------------------------------------------------------------------------
// Fragment slot mapping for pre-packed B operands.
// Logical B[k][n] (K x N) is stored as tiles: tile (tn,tk) holds a 32x16 block
// in contiguous-k b-operand lane order (same as A): lane = (k%32>=16 ? 16:0) + (n%16), h = k%16.
// Each lane's 16 halves are contiguous -> global_load_b128 x2 per wmma.
// ---------------------------------------------------------------------------
__device__ __forceinline__ long bslot(int k, int n, int ntk) {
    int tk = k >> 5, kr = k & 31;
    int tn = n >> 4;
    int lane = ((kr >= 16) ? 16 : 0) + (n & 15);
    return ((long)(tn * ntk + tk) * 32 + lane) * 16 + (kr & 15);
}

// pack: logical B[k][coff+c] = src[k*C + c]   (src is [R,C] f32, K=R)
__global__ __launch_bounds__(256)
void k_packB_frag(_Float16* __restrict__ dst, const float* __restrict__ src,
                  int R, int C, int coff, int ntk) {
    int i = blockIdx.x * 256 + threadIdx.x;
    if (i >= R * C) return;
    int k = i / C, c = i % C;
    dst[bslot(k, coff + c, ntk)] = (_Float16)src[i];
}
// packT: logical B[k][coff+r] = src[r*C + k]  (src is [R,C] f32, K=C, N-block=R)
__global__ __launch_bounds__(256)
void k_packBT_frag(_Float16* __restrict__ dst, const float* __restrict__ src,
                   int R, int C, int coff, int ntk) {
    int i = blockIdx.x * 256 + threadIdx.x;
    if (i >= R * C) return;
    int r = i / C, k = i % C;
    dst[bslot(k, coff + r, ntk)] = (_Float16)src[i];
}

// ---------------------------------------------------------------------------
// WMMA GEMM: C[M,N] = A[M,K](f32 -> f16) * Bp(fragment-packed f16) + bias[N]
// Block = 128 threads (4 waves). C-tile 32 x 64: each wave one 16-col strip,
// two 16x16 accumulators. M must be a multiple of 16 (true for all callers).
// ---------------------------------------------------------------------------
__global__ __launch_bounds__(128)
void k_gemm(const float* __restrict__ A, const _Float16* __restrict__ Bp,
            const float* __restrict__ bias, float* __restrict__ C,
            int M, int N, int K, int ntk, int ntn)
{
    __shared__ _Float16 Af[2 * 512];           // two 16x32 A tiles, fragment order
    const int tid  = threadIdx.x;
    const int lane = tid & 31;
    const int wave = tid >> 5;
    const int row0 = blockIdx.x * 32;
    const int tn   = blockIdx.y * 4 + wave;
    const int tnc  = (tn < ntn) ? tn : (ntn - 1);   // clamp so EXEC stays full
    // producer role: thread writes Af[g*512 + lane*16 + wave*4 .. +3]
    const int arow  = lane & 15;
    const int kbase = (lane < 16) ? 0 : 16;
    const int ks    = kbase + 4 * wave; // 0,4,8,12
    const bool g1ok = (row0 + 16) < M;              // second row-group valid
    const float* A0 = A + (long)(row0 + arow) * K;
    const float* A1 = A + (long)min(row0 + 16 + arow, M - 1) * K;
    _Float16* af0 = Af + lane * 16 + wave * 4;
    _Float16* af1 = af0 + 512;

    v8f acc0 = {}, acc1 = {};
    const int kfull = K & ~31;
    int tk = 0;
    for (int kk = 0; kk < kfull; kk += 32, ++tk) {
        f32x2 a0 = *(const f32x2*)(A0 + kk + ks);
        f32x2 a1 = *(const f32x2*)(A0 + kk + ks + 2);
        f32x2 b0 = *(const f32x2*)(A1 + kk + ks);
        f32x2 b1 = *(const f32x2*)(A1 + kk + ks + 2);
        v4h p0 = {(_Float16)a0.x, (_Float16)a0.y, (_Float16)a1.x, (_Float16)a1.y};
        v4h p1 = {(_Float16)b0.x, (_Float16)b0.y, (_Float16)b1.x, (_Float16)b1.y};
        *(v4h*)af0 = p0;
        *(v4h*)af1 = p1;
        __syncthreads();
        v16h av0 = *(const v16h*)(Af + lane * 16);
        v16h av1 = *(const v16h*)(Af + 512 + lane * 16);
        v16h bv  = *(const v16h*)(Bp + ((long)(tnc * ntk + tk) * 32 + lane) * 16);
        acc0 = __builtin_amdgcn_wmma_f32_16x16x32_f16(false, av0, false, bv,
                                                      (short)0, acc0, false, false);
        acc1 = __builtin_amdgcn_wmma_f32_16x16x32_f16(false, av1, false, bv,
                                                      (short)0, acc1, false, false);
        __syncthreads();
    }
    if (kfull < K) {                                // branchless clamped tail
        _Float16 t0[4], t1[4];
        #pragma unroll
        for (int i = 0; i < 4; ++i) {
            int gk = kfull + ks + i;
            int ck = min(gk, K - 1);
            float v0 = A0[ck], v1 = A1[ck];
            bool ok = gk < K;
            t0[i] = (_Float16)(ok ? v0 : 0.f);
            t1[i] = (_Float16)(ok ? v1 : 0.f);
        }
        v4h p0 = {t0[0], t0[1], t0[2], t0[3]};
        v4h p1 = {t1[0], t1[1], t1[2], t1[3]};
        *(v4h*)af0 = p0;
        *(v4h*)af1 = p1;
        __syncthreads();
        v16h av0 = *(const v16h*)(Af + lane * 16);
        v16h av1 = *(const v16h*)(Af + 512 + lane * 16);
        v16h bv  = *(const v16h*)(Bp + ((long)(tnc * ntk + tk) * 32 + lane) * 16);
        acc0 = __builtin_amdgcn_wmma_f32_16x16x32_f16(false, av0, false, bv,
                                                      (short)0, acc0, false, false);
        acc1 = __builtin_amdgcn_wmma_f32_16x16x32_f16(false, av1, false, bv,
                                                      (short)0, acc1, false, false);
    }
    if (tn < ntn) {
        int gn = tnc * 16 + (lane & 15);
        if (gn < N) {
            float bs = bias[gn];
            #pragma unroll
            for (int r = 0; r < 8; ++r) {
                int m = r + ((lane >= 16) ? 8 : 0);
                C[(long)(row0 + m) * N + gn] = acc0[r] + bs;
                if (g1ok) C[(long)(row0 + 16 + m) * N + gn] = acc1[r] + bs;
            }
        }
    }
}

// ---------------------------------------------------------------------------
// Utility kernels
// ---------------------------------------------------------------------------
__global__ __launch_bounds__(256) void k_fill_f32(float* p, long n, float v) {
    long i = (long)blockIdx.x * 256 + threadIdx.x;
    if (i < n) p[i] = v;
}
__global__ __launch_bounds__(256) void k_fill_u32(unsigned* p, long n, unsigned v) {
    long i = (long)blockIdx.x * 256 + threadIdx.x;
    if (i < n) p[i] = v;
}
__global__ void k_copy_f32(float* dst, const float* src, int n, int off) {
    int i = threadIdx.x;
    if (i < n) dst[off + i] = src[i];
}

// ---------------------------------------------------------------------------
// GATv2 edge kernels (scatter softmax over dst, 4 heads x 16 channels)
// ---------------------------------------------------------------------------
__device__ __forceinline__ unsigned f32key(float f) {
    unsigned b = __float_as_uint(f);
    return (b & 0x80000000u) ? ~b : (b | 0x80000000u);
}
__device__ __forceinline__ float keyf32(unsigned k) {
    unsigned b = (k & 0x80000000u) ? (k & 0x7FFFFFFFu) : ~k;
    return __uint_as_float(b);
}

__global__ __launch_bounds__(256)
void k_gat_logit_max(const float* __restrict__ XP, int ldx, int loff, int roff,
                     const int* __restrict__ src, const int* __restrict__ dst,
                     int Ereal, int Etot, const float* __restrict__ att,
                     float* __restrict__ logit, unsigned* __restrict__ mkey)
{
    int idx = blockIdx.x * 256 + threadIdx.x;
    if (idx >= Etot * 4) return;
    int e = idx >> 2, h = idx & 3;
    int s = (e < Ereal) ? src[e] : (e - Ereal);
    int d = (e < Ereal) ? dst[e] : (e - Ereal);
    const float4* xs4 = (const float4*)(XP + (long)s * ldx + loff + h * 16);
    const float4* xd4 = (const float4*)(XP + (long)d * ldx + roff + h * 16);
    const float4* a4  = (const float4*)(att + h * 16);
    float acc = 0.f;
    #pragma unroll
    for (int i = 0; i < 4; ++i) {
        float4 u = xs4[i], w = xd4[i], a = a4[i];
        float v;
        v = u.x + w.x; v = (v >= 0.f) ? v : 0.2f * v; acc += v * a.x;
        v = u.y + w.y; v = (v >= 0.f) ? v : 0.2f * v; acc += v * a.y;
        v = u.z + w.z; v = (v >= 0.f) ? v : 0.2f * v; acc += v * a.z;
        v = u.w + w.w; v = (v >= 0.f) ? v : 0.2f * v; acc += v * a.w;
    }
    logit[idx] = acc;
    atomicMax(&mkey[d * 4 + h], f32key(acc));
}

__global__ __launch_bounds__(256)
void k_gat_expsum(const int* __restrict__ src, const int* __restrict__ dst,
                  int Ereal, int Etot, const float* __restrict__ logit,
                  const unsigned* __restrict__ mkey,
                  float* __restrict__ ex, float* __restrict__ den)
{
    int idx = blockIdx.x * 256 + threadIdx.x;
    if (idx >= Etot * 4) return;
    int e = idx >> 2, h = idx & 3;
    int d = (e < Ereal) ? dst[e] : (e - Ereal);
    float m = keyf32(mkey[d * 4 + h]);
    float x = __expf(logit[idx] - m);
    ex[idx] = x;
    atomicAdd(&den[d * 4 + h], x);
}

__global__ __launch_bounds__(256)
void k_gat_agg(const float* __restrict__ XP, int ldx, int loff,
               const int* __restrict__ src, const int* __restrict__ dst,
               int Ereal, int Etot, const float* __restrict__ ex,
               const float* __restrict__ den, float* __restrict__ agg)
{
    int idx = blockIdx.x * 256 + threadIdx.x;
    if (idx >= Etot * 4) return;
    int e = idx >> 2, h = idx & 3;
    int s = (e < Ereal) ? src[e] : (e - Ereal);
    int d = (e < Ereal) ? dst[e] : (e - Ereal);
    float alpha = ex[idx] / den[d * 4 + h];
    const float4* xs4 = (const float4*)(XP + (long)s * ldx + loff + h * 16);
    float* ag = agg + (long)d * 64 + h * 16;
    #pragma unroll
    for (int i = 0; i < 4; ++i) {
        float4 u = xs4[i];
        atomicAdd(&ag[4 * i + 0], alpha * u.x);
        atomicAdd(&ag[4 * i + 1], alpha * u.y);
        atomicAdd(&ag[4 * i + 2], alpha * u.z);
        atomicAdd(&ag[4 * i + 3], alpha * u.w);
    }
}

__global__ __launch_bounds__(256)
void k_gat_finish(const float* __restrict__ agg, const float* __restrict__ XP,
                  int ldx, int resoff, float* __restrict__ out, int n_nodes)
{
    int idx = blockIdx.x * 256 + threadIdx.x;
    if (idx >= n_nodes * 16) return;
    int n = idx >> 4, c = idx & 15;
    const float* ag = agg + (long)n * 64;
    out[idx] = 0.25f * (ag[c] + ag[16 + c] + ag[32 + c] + ag[48 + c])
             + XP[(long)n * ldx + resoff + c];
}

// ---------------------------------------------------------------------------
// BatchNorm (training-mode batch stats) + optional PReLU / leaky
// ---------------------------------------------------------------------------
__global__ __launch_bounds__(128)
void k_bn_stats(const float* __restrict__ X, int M, int C, float* __restrict__ st)
{
    int c = threadIdx.x;
    if (c >= C) return;
    float s = 0.f, q = 0.f;
    for (int r = blockIdx.x; r < M; r += gridDim.x) {
        float v = X[(long)r * C + c];
        s += v; q += v * v;
    }
    atomicAdd(&st[c], s);
    atomicAdd(&st[C + c], q);
}

__global__ __launch_bounds__(256)
void k_bn_apply(const float* __restrict__ X, float* __restrict__ Y, int M, int C,
                const float* __restrict__ st, const float* __restrict__ g,
                const float* __restrict__ b, const float* __restrict__ slope, int act)
{
    long idx = (long)blockIdx.x * 256 + threadIdx.x;
    if (idx >= (long)M * C) return;
    int c = (int)(idx % C);
    float mean = st[c] / M;
    float var  = st[C + c] / M - mean * mean;
    float y = g[c] * (X[idx] - mean) * rsqrtf(var + 1e-5f) + b[c];
    if (act == 1 && y < 0.f) y *= slope[c];
    if (act == 2 && y < 0.f) y *= 0.01f;
    Y[idx] = y;
}

// ---------------------------------------------------------------------------
// Misc pipeline kernels
// ---------------------------------------------------------------------------
__global__ __launch_bounds__(256)
void k_concat32(const float* __restrict__ X1, const float* __restrict__ X2,
                float* __restrict__ XA, int n_nodes)
{
    int idx = blockIdx.x * 256 + threadIdx.x;
    if (idx >= n_nodes * 32) return;
    int n = idx >> 5, c = idx & 31;
    XA[idx] = (c < 16) ? X1[n * 16 + c] : X2[n * 16 + (c - 16)];
}

__global__ __launch_bounds__(256)
void k_ef_build(const float* __restrict__ XA, const int* __restrict__ lsrc,
                const int* __restrict__ ldst, const float* __restrict__ efeat,
                float* __restrict__ EF, int Lrows)
{
    long idx = (long)blockIdx.x * 256 + threadIdx.x;
    if (idx >= (long)Lrows * 86) return;
    int l = (int)(idx / 86), j = (int)(idx % 86);
    float v;
    if (j < 32)      v = XA[(long)lsrc[l] * 32 + j];
    else if (j < 64) v = XA[(long)ldst[l] * 32 + (j - 32)];
    else             v = efeat[(long)l * 22 + (j - 64)];
    EF[idx] = v;
}

__device__ __forceinline__ float sigm(float x) { return 1.f / (1.f + __expf(-x)); }

__global__ __launch_bounds__(256)
void k_gru_gate(const float* __restrict__ GI, const float* __restrict__ GH,
                float* __restrict__ H, float* __restrict__ Y, int Lrows)
{
    int idx = blockIdx.x * 256 + threadIdx.x;
    if (idx >= Lrows * 32) return;
    int l = idx >> 5, j = idx & 31;
    const float* gi = GI + (long)l * 96;
    const float* gh = GH + (long)l * 96;
    float r  = sigm(gi[j]      + gh[j]);
    float z  = sigm(gi[32 + j] + gh[32 + j]);
    float nn = tanhf(gi[64 + j] + r * gh[64 + j]);
    float h  = H[idx];
    float hn = (1.f - z) * nn + z * h;
    H[idx] = hn;
    Y[(long)l * 64 + j] = hn;
}

__global__ __launch_bounds__(256)
void k_time_agg(const float* __restrict__ Y2, const float* __restrict__ W,
                const float* __restrict__ b, float* __restrict__ out, int Lrows)
{
    int idx = blockIdx.x * 256 + threadIdx.x;
    if (idx >= Lrows * 64) return;
    int l = idx >> 6, c = idx & 63;
    float s = b[0];
    #pragma unroll
    for (int t = 0; t < 5; ++t) s += Y2[((long)t * Lrows + l) * 64 + c] * W[t];
    out[idx] = s;
}

// ---------------------------------------------------------------------------
// Host orchestration
// ---------------------------------------------------------------------------
extern "C" void kernel_launch(void* const* d_in, const int* in_sizes, int n_in,
                              void* d_out, int out_size, void* d_ws, size_t ws_size,
                              hipStream_t stream)
{
    (void)in_sizes; (void)out_size; (void)ws_size;
    const int N = 50000, E = 800000, L = 100000, T = 5;
    const int ET = E + N;

    const float* x_per_year = (const float*)d_in[0];   // [T,N,128]
    const int*   lei        = (const int*)d_in[1];     // [2,L]
    const int*   eiy        = (const int*)d_in[2];     // [T,2,E]
    const float* lef        = (const float*)d_in[3];   // [T,L,22]

    const float* P[52];
    static const int psz[52] = {
        5,1, 16,16, 16,16,
        8192,8192,2048,64,16,64,64,
        1024,1024,256,64,16,64,64,
        3072,8256,96,96, 3072,8256,96,96,
        3072,6144,96,96, 3072,6144,96,96,
        86,86,
        64,64, 32,32, 8,8,
        2048,32, 256,8, 16,2,
        16,16 };
    if (n_in >= 4 + 52) {
        for (int i = 0; i < 52; ++i) P[i] = (const float*)d_in[4 + i];
    } else {
        const float* base = (const float*)d_in[4];
        for (int i = 0; i < 52; ++i) { P[i] = base; base += psz[i]; }
    }
    const float *aggW = P[0], *aggB = P[1];
    const float *bn1b = P[2], *bn1g = P[3], *bn2b = P[4], *bn2g = P[5];
    const float *g1Wl = P[6], *g1Wr = P[7], *g1Wres = P[8], *g1att = P[9],
                *g1bias = P[10], *g1bl = P[11], *g1br = P[12];
    const float *g2Wl = P[13], *g2Wr = P[14], *g2Wres = P[15], *g2att = P[16],
                *g2bias = P[17], *g2bl = P[18], *g2br = P[19];
    const float* gru_w[2][2][4];
    for (int l = 0; l < 2; ++l)
        for (int d = 0; d < 2; ++d)
            for (int k = 0; k < 4; ++k)
                gru_w[l][d][k] = P[20 + l * 8 + d * 4 + k];
    const float *lnb = P[36], *lng = P[37];
    const float *bn0b_m = P[38], *bn0g_m = P[39], *bn1b_m = P[40], *bn1g_m = P[41],
                *bn2b_m = P[42], *bn2g_m = P[43];
    const float *fc1W = P[44], *fc1b = P[45], *fc2W = P[46], *fc2b = P[47],
                *fc3W = P[48], *fc3b = P[49], *prelu1 = P[50], *prelu2 = P[51];

    // ---- workspace carve ----
    char* ws = (char*)d_ws;
    size_t off = 0;
    auto carve = [&](size_t bytes) -> char* {
        char* p = ws + off; off += (bytes + 255) & ~(size_t)255; return p;
    };
    float*    Y0 = (float*)carve((size_t)T * L * 86 * 4);
    float*    Y1 = (float*)carve((size_t)T * L * 64 * 4);
    // fragment-packed f16 B operands (sizes = ntn*ntk*512 halves)
    _Float16* Bcat1 = (_Float16*)carve((size_t)9 * 4 * 512 * 2);  // K=128,N=144
    _Float16* Bcat2 = (_Float16*)carve((size_t)9 * 1 * 512 * 2);  // K=16, N=144
    float*    biascat1 = (float*)carve(144 * 4);
    float*    biascat2 = (float*)carve(144 * 4);
    _Float16* WihT[2][2]; _Float16* WhhT[2][2];
    for (int l = 0; l < 2; ++l)
        for (int d = 0; d < 2; ++d) {
            WihT[l][d] = (_Float16*)carve((size_t)6 * 3 * 512 * 2); // K<=86,N=96
            WhhT[l][d] = (_Float16*)carve((size_t)6 * 1 * 512 * 2); // K=32, N=96
        }
    _Float16* fc1T = (_Float16*)carve((size_t)2 * 2 * 512 * 2);     // K=64,N=32
    _Float16* fc2T = (_Float16*)carve((size_t)1 * 1 * 512 * 2);     // K=32,N=8
    _Float16* fc3T = (_Float16*)carve((size_t)1 * 1 * 512 * 2);     // K=8, N=2
    float*    bnst = (float*)carve(2 * 128 * 4);
    size_t scratch_base = off;

    // Phase A arena (GAT)
    float*    XP   = (float*)carve((size_t)N * 144 * 4);
    float*    XP2  = (float*)carve((size_t)N * 144 * 4);
    float*    X1   = (float*)carve((size_t)N * 16 * 4);
    float*    X2   = (float*)carve((size_t)N * 16 * 4);
    float*    XA   = (float*)carve((size_t)N * 32 * 4);
    float*    logit= (float*)carve((size_t)ET * 4 * 4);
    float*    exv  = (float*)carve((size_t)ET * 4 * 4);
    unsigned* mkey = (unsigned*)carve((size_t)N * 4 * 4);
    float*    den  = (float*)carve((size_t)N * 4 * 4);
    float*    aggbuf = (float*)carve((size_t)N * 64 * 4);

    // Phase B arena aliases phase A
    off = scratch_base;
    float* GI  = (float*)carve((size_t)T * L * 96 * 4);
    float* GH  = (float*)carve((size_t)L * 96 * 4);
    float* Hb  = (float*)carve((size_t)L * 32 * 4);
    float* Y2  = (float*)carve((size_t)T * L * 64 * 4);
    float* H64 = (float*)carve((size_t)L * 64 * 4);
    float* H32 = (float*)carve((size_t)L * 32 * 4);
    float* H8  = (float*)carve((size_t)L * 8 * 4);

    // ---- pack weights into fragment order (zero-fill pads first) ----
    k_fill_u32<<<(int)cdivl(9*4*256,256),256,0,stream>>>((unsigned*)Bcat1, 9*4*256, 0u);
    k_packB_frag<<<(int)cdivl(128*64,256),256,0,stream>>>(Bcat1, g1Wl, 128, 64, 0, 4);
    k_packB_frag<<<(int)cdivl(128*64,256),256,0,stream>>>(Bcat1, g1Wr, 128, 64, 64, 4);
    k_packB_frag<<<(int)cdivl(128*16,256),256,0,stream>>>(Bcat1, g1Wres, 128, 16, 128, 4);
    k_copy_f32<<<1,64,0,stream>>>(biascat1, g1bl, 64, 0);
    k_copy_f32<<<1,64,0,stream>>>(biascat1, g1br, 64, 64);
    k_copy_f32<<<1,16,0,stream>>>(biascat1, g1bias, 16, 128);
    k_fill_u32<<<(int)cdivl(9*256,256),256,0,stream>>>((unsigned*)Bcat2, 9*256, 0u);
    k_packB_frag<<<(int)cdivl(16*64,256),256,0,stream>>>(Bcat2, g2Wl, 16, 64, 0, 1);
    k_packB_frag<<<(int)cdivl(16*64,256),256,0,stream>>>(Bcat2, g2Wr, 16, 64, 64, 1);
    k_packB_frag<<<(int)cdivl(16*16,256),256,0,stream>>>(Bcat2, g2Wres, 16, 16, 128, 1);
    k_copy_f32<<<1,64,0,stream>>>(biascat2, g2bl, 64, 0);
    k_copy_f32<<<1,64,0,stream>>>(biascat2, g2br, 64, 64);
    k_copy_f32<<<1,16,0,stream>>>(biascat2, g2bias, 16, 128);
    for (int l = 0; l < 2; ++l)
        for (int d = 0; d < 2; ++d) {
            int kin = (l == 0) ? 86 : 64;
            int ntk = (kin + 31) / 32;                 // 3 or 2
            k_fill_u32<<<(int)cdivl(6*3*256,256),256,0,stream>>>((unsigned*)WihT[l][d], 6*3*256, 0u);
            k_packBT_frag<<<(int)cdivl(96*kin,256),256,0,stream>>>(WihT[l][d], gru_w[l][d][1], 96, kin, 0, ntk);
            k_fill_u32<<<(int)cdivl(6*256,256),256,0,stream>>>((unsigned*)WhhT[l][d], 6*256, 0u);
            k_packBT_frag<<<(int)cdivl(96*32,256),256,0,stream>>>(WhhT[l][d], gru_w[l][d][0], 96, 32, 0, 1);
        }
    k_fill_u32<<<(int)cdivl(2*2*256,256),256,0,stream>>>((unsigned*)fc1T, 2*2*256, 0u);
    k_packBT_frag<<<(int)cdivl(32*64,256),256,0,stream>>>(fc1T, fc1W, 32, 64, 0, 2);
    k_fill_u32<<<1,256,0,stream>>>((unsigned*)fc2T, 256, 0u);
    k_packBT_frag<<<1,256,0,stream>>>(fc2T, fc2W, 8, 32, 0, 1);
    k_fill_u32<<<1,256,0,stream>>>((unsigned*)fc3T, 256, 0u);
    k_packBT_frag<<<1,256,0,stream>>>(fc3T, fc3W, 2, 8, 0, 1);

    // ---- Phase A: per-timestep GNN ----
    for (int t = 0; t < T; ++t) {
        const float* Xt   = x_per_year + (size_t)t * N * 128;
        const int*   srcT = eiy + (size_t)t * 2 * E;
        const int*   dstT = srcT + E;

        k_gemm<<<dim3((unsigned)cdivl(N,32), 3), 128, 0, stream>>>(Xt, Bcat1, biascat1, XP, N, 144, 128, 4, 9);
        k_fill_u32<<<(int)cdivl((long)N*4,256),256,0,stream>>>(mkey, (long)N*4, 0u);
        k_fill_f32<<<(int)cdivl((long)N*4,256),256,0,stream>>>(den, (long)N*4, 0.f);
        k_fill_f32<<<(int)cdivl((long)N*64,256),256,0,stream>>>(aggbuf, (long)N*64, 0.f);
        k_gat_logit_max<<<(int)cdivl((long)ET*4,256),256,0,stream>>>(XP,144,0,64, srcT,dstT,E,ET, g1att, logit, mkey);
        k_gat_expsum  <<<(int)cdivl((long)ET*4,256),256,0,stream>>>(srcT,dstT,E,ET, logit, mkey, exv, den);
        k_gat_agg     <<<(int)cdivl((long)ET*4,256),256,0,stream>>>(XP,144,0, srcT,dstT,E,ET, exv, den, aggbuf);
        k_gat_finish  <<<(int)cdivl((long)N*16,256),256,0,stream>>>(aggbuf, XP, 144, 128, X1, N);
        k_fill_f32<<<1,256,0,stream>>>(bnst, 256, 0.f);
        k_bn_stats<<<256,128,0,stream>>>(X1, N, 16, bnst);
        k_bn_apply<<<(int)cdivl((long)N*16,256),256,0,stream>>>(X1, X1, N, 16, bnst, bn1g, bn1b, prelu1, 1);

        k_gemm<<<dim3((unsigned)cdivl(N,32), 3), 128, 0, stream>>>(X1, Bcat2, biascat2, XP2, N, 144, 16, 1, 9);
        k_fill_u32<<<(int)cdivl((long)N*4,256),256,0,stream>>>(mkey, (long)N*4, 0u);
        k_fill_f32<<<(int)cdivl((long)N*4,256),256,0,stream>>>(den, (long)N*4, 0.f);
        k_fill_f32<<<(int)cdivl((long)N*64,256),256,0,stream>>>(aggbuf, (long)N*64, 0.f);
        k_gat_logit_max<<<(int)cdivl((long)ET*4,256),256,0,stream>>>(XP2,144,0,64, srcT,dstT,E,ET, g2att, logit, mkey);
        k_gat_expsum  <<<(int)cdivl((long)ET*4,256),256,0,stream>>>(srcT,dstT,E,ET, logit, mkey, exv, den);
        k_gat_agg     <<<(int)cdivl((long)ET*4,256),256,0,stream>>>(XP2,144,0, srcT,dstT,E,ET, exv, den, aggbuf);
        k_gat_finish  <<<(int)cdivl((long)N*16,256),256,0,stream>>>(aggbuf, XP2, 144, 128, X2, N);
        k_fill_f32<<<1,256,0,stream>>>(bnst, 256, 0.f);
        k_bn_stats<<<256,128,0,stream>>>(X2, N, 16, bnst);
        k_bn_apply<<<(int)cdivl((long)N*16,256),256,0,stream>>>(X2, X2, N, 16, bnst, bn2g, bn2b, prelu2, 1);

        k_concat32<<<(int)cdivl((long)N*32,256),256,0,stream>>>(X1, X2, XA, N);

        float* EF = Y0 + (size_t)t * L * 86;
        k_ef_build<<<(int)cdivl((long)L*86,256),256,0,stream>>>(XA, lei, lei + L, lef + (size_t)t * L * 22, EF, L);
        k_fill_f32<<<1,256,0,stream>>>(bnst, 256, 0.f);
        k_bn_stats<<<256,128,0,stream>>>(EF, L, 86, bnst);
        k_bn_apply<<<(int)cdivl((long)L*86,256),256,0,stream>>>(EF, EF, L, 86, bnst, lng, lnb, nullptr, 0);
    }

    // ---- Phase B: bidirectional GRU x2 ----
    for (int layer = 0; layer < 2; ++layer) {
        const float* Xin = (layer == 0) ? Y0 : Y1;
        float*       Yo  = (layer == 0) ? Y1 : Y2;
        int kin = (layer == 0) ? 86 : 64;
        int ntk = (kin + 31) / 32;
        for (int dir = 0; dir < 2; ++dir) {
            k_gemm<<<dim3((unsigned)cdivl((long)T*L,32), 2), 128, 0, stream>>>(
                Xin, WihT[layer][dir], gru_w[layer][dir][3], GI, T * L, 96, kin, ntk, 6);
            k_fill_f32<<<(int)cdivl((long)L*32,256),256,0,stream>>>(Hb, (long)L*32, 0.f);
            for (int s = 0; s < T; ++s) {
                int t = dir ? (T - 1 - s) : s;
                k_gemm<<<dim3((unsigned)cdivl(L,32), 2), 128, 0, stream>>>(
                    Hb, WhhT[layer][dir], gru_w[layer][dir][2], GH, L, 96, 32, 1, 6);
                k_gru_gate<<<(int)cdivl((long)L*32,256),256,0,stream>>>(
                    GI + (size_t)t * L * 96, GH, Hb, Yo + (size_t)t * L * 64 + dir * 32, L);
            }
        }
    }

    // ---- time aggregation -> x_feat ----
    float* xfeat = (float*)d_out;
    float* outp  = (float*)d_out + (size_t)L * 64;
    k_time_agg<<<(int)cdivl((long)L*64,256),256,0,stream>>>(Y2, aggW, aggB, xfeat, L);

    // ---- MLP head ----
    k_fill_f32<<<1,256,0,stream>>>(bnst, 256, 0.f);
    k_bn_stats<<<256,128,0,stream>>>(xfeat, L, 64, bnst);
    k_bn_apply<<<(int)cdivl((long)L*64,256),256,0,stream>>>(xfeat, H64, L, 64, bnst, bn0g_m, bn0b_m, nullptr, 0);
    k_gemm<<<dim3((unsigned)cdivl(L,32), 1), 128, 0, stream>>>(H64, fc1T, fc1b, H32, L, 32, 64, 2, 2);
    k_fill_f32<<<1,256,0,stream>>>(bnst, 256, 0.f);
    k_bn_stats<<<256,128,0,stream>>>(H32, L, 32, bnst);
    k_bn_apply<<<(int)cdivl((long)L*32,256),256,0,stream>>>(H32, H32, L, 32, bnst, bn1g_m, bn1b_m, nullptr, 2);
    k_gemm<<<dim3((unsigned)cdivl(L,32), 1), 128, 0, stream>>>(H32, fc2T, fc2b, H8, L, 8, 32, 1, 1);
    k_fill_f32<<<1,256,0,stream>>>(bnst, 256, 0.f);
    k_bn_stats<<<256,128,0,stream>>>(H8, L, 8, bnst);
    k_bn_apply<<<(int)cdivl((long)L*8,256),256,0,stream>>>(H8, H8, L, 8, bnst, bn2g_m, bn2b_m, nullptr, 2);
    k_gemm<<<dim3((unsigned)cdivl(L,32), 1), 128, 0, stream>>>(H8, fc3T, fc3b, outp, L, 2, 8, 1, 1);
}
